// RelevanceModel_36086315221589
// MI455X (gfx1250) — matrix operand order repaired and checked by reading, not compiled, not run on hardware
//
#include <hip/hip_runtime.h>
#include <stdint.h>

typedef __attribute__((ext_vector_type(16))) __bf16 v16bf;
typedef __attribute__((ext_vector_type(8)))  float  v8f;

#define T_DIST    0.2f
#define MASK_FILL -1e10f
#define COS_EPS   1e-8f

static __device__ __forceinline__ unsigned short f2bf(float f) {
    union { float f; uint32_t u; } c; c.f = f;
    uint32_t u = c.u;
    uint32_t r = u + 0x7FFFu + ((u >> 16) & 1u);   // round-to-nearest-even
    return (unsigned short)(r >> 16);
}

union Frag {
    uint4 u[2];
    v16bf v;
};

// A fragment from LDS: lane holds row (lane&15); halfs e=0..7 -> K=kbase+0..7,
// e=8..15 -> K=kbase+16..23  => two 16B LDS loads 32B apart.
static __device__ __forceinline__ void loadA(Frag& f,
                                             const unsigned short* base, int ks) {
    const uint4* p = (const uint4*)(base + ks * 32);
    f.u[0] = p[0];
    f.u[1] = p[2];     // +16 halfs = +32 bytes
}

// B fragment from global WkT: lane holds col N (lane&15); halfs e=0..7 -> K=kbase+0..7,
// e=8..15 -> K=kbase+16..23  => two 16B global loads 32B apart.
static __device__ __forceinline__ void loadB(Frag& f,
                                             const unsigned short* base,
                                             int ks, int H) {
    const uint4* p = (const uint4*)(base + (size_t)ks * 32);
    f.u[0] = p[0];
    f.u[1] = p[2];
}

// ---------------------------------------------------------------------------
// Kernel 1: WkT[j][h] = bf16(Wk[j][h])  (coalesced writes, L2-resident output)
// ---------------------------------------------------------------------------
__global__ void wkt_convert_kernel(const float* __restrict__ Wk,
                                   unsigned short* __restrict__ WkT, int H) {
    int id = blockIdx.x * blockDim.x + threadIdx.x;
    int total = H * H;
    if (id < total) {
        int h = id / H;
        int j = id - h * H;
        WkT[id] = f2bf(Wk[h * H + j]);
    }
}

// ---------------------------------------------------------------------------
// Kernel 2: q[j] = sum_h cu[h] * Wq[j][h]
// ---------------------------------------------------------------------------
__global__ void qproj_kernel(const float* __restrict__ cu,
                             const float* __restrict__ Wq,
                             float* __restrict__ q, int H) {
    int j = blockIdx.x * blockDim.x + threadIdx.x;
    if (j < H) {
        const float* wr = Wq + (size_t)j * H;
        float acc = 0.f;
        for (int h = 0; h < H; ++h) acc += cu[h] * wr[h];
        q[j] = acc;
    }
}

// ---------------------------------------------------------------------------
// Kernel 3: bf16 WMMA k-projection with fused per-row reductions.
// Block = 256 threads = 8 waves; each wave owns a 16-row tile of memory.
// Two N-tiles per pass (shared A, two acc chains), 2-deep ks pipeline.
// Outputs per row i: dot[i] = k_i . q   and   ss[i] = ||k_i||^2
// ---------------------------------------------------------------------------
__global__ void __launch_bounds__(256, 1)
kproj_wmma_kernel(const float* __restrict__ mem,
                  const unsigned short* __restrict__ WkT,
                  const float* __restrict__ qv,
                  float* __restrict__ dotv, float* __restrict__ ssv,
                  int W, int H) {
    extern __shared__ unsigned char smem[];
    float* q_lds = (float*)smem;                                   // H floats
    unsigned short* A = (unsigned short*)(smem + (size_t)H * 4);   // 128*H bf16

    const int tid = threadIdx.x;

    // stage q
    for (int i = tid; i < H; i += 256) q_lds[i] = qv[i];

    // stage 128 rows of memory as bf16 into LDS (fully coalesced float4 loads)
    const long rowBase = (long)blockIdx.x * 128;
    const float4* gsrc = (const float4*)(mem + rowBase * H);
    const int nf4 = (128 * H) >> 2;
    for (int l = tid; l < nf4; l += 256) {
        float4 f = gsrc[l];
        uint2 p;
        p.x = (uint32_t)f2bf(f.x) | ((uint32_t)f2bf(f.y) << 16);
        p.y = (uint32_t)f2bf(f.z) | ((uint32_t)f2bf(f.w) << 16);
        ((uint2*)A)[l] = p;
    }
    __syncthreads();

    const int wave = tid >> 5;
    const int lane = tid & 31;
    const int lrow  = (wave << 4) + (lane & 15);   // A row this lane carries
    const int kbase = (lane & 16) >> 1;            // 0 or 8 (A layout half-split)
    const unsigned short* Abase = A + (size_t)lrow * H + kbase;

    float pdot[8], pss[8];
#pragma unroll
    for (int r = 0; r < 8; ++r) { pdot[r] = 0.f; pss[r] = 0.f; }

    const int nJT = H >> 4;   // 16-col tiles of k (even; processed in pairs)
    const int nKS = H >> 5;   // 32-deep K steps (even)

    for (int jt = 0; jt < nJT; jt += 2) {
        v8f acc0 = {0.f, 0.f, 0.f, 0.f, 0.f, 0.f, 0.f, 0.f};
        v8f acc1 = {0.f, 0.f, 0.f, 0.f, 0.f, 0.f, 0.f, 0.f};

        const unsigned short* B0base = WkT + (size_t)(jt * 16 + (lane & 15)) * H + kbase;
        const unsigned short* B1base = B0base + (size_t)16 * H;

        Frag a0, b00, b10, a1, b01, b11;
        loadA(a0, Abase, 0);
        loadB(b00, B0base, 0, H);
        loadB(b10, B1base, 0, H);

        for (int ks = 0; ks < nKS; ks += 2) {
            // prefetch ks+1 into the second buffer while ks computes
            loadA(a1, Abase, ks + 1);
            loadB(b01, B0base, ks + 1, H);
            loadB(b11, B1base, ks + 1, H);

            acc0 = __builtin_amdgcn_wmma_f32_16x16x32_bf16(
                false, a0.v, false, b00.v, (short)0, acc0, false, false);
            acc1 = __builtin_amdgcn_wmma_f32_16x16x32_bf16(
                false, a0.v, false, b10.v, (short)0, acc1, false, false);

            if (ks + 2 < nKS) {
                // prefetch ks+2 into the first buffer while ks+1 computes
                loadA(a0, Abase, ks + 2);
                loadB(b00, B0base, ks + 2, H);
                loadB(b10, B1base, ks + 2, H);
            }

            acc0 = __builtin_amdgcn_wmma_f32_16x16x32_bf16(
                false, a1.v, false, b01.v, (short)0, acc0, false, false);
            acc1 = __builtin_amdgcn_wmma_f32_16x16x32_bf16(
                false, a1.v, false, b11.v, (short)0, acc1, false, false);
        }

        // fold C tiles: lane covers col (jt|jt+1)*16+(lane&15); VGPR r -> row
        float qj0 = q_lds[jt * 16 + (lane & 15)];
        float qj1 = q_lds[jt * 16 + 16 + (lane & 15)];
#pragma unroll
        for (int r = 0; r < 8; ++r) {
            float c0 = acc0[r];
            float c1 = acc1[r];
            pdot[r] += c0 * qj0 + c1 * qj1;
            pss[r]  += c0 * c0 + c1 * c1;
        }
    }

    // reduce across the 16 lanes sharing each row group
#pragma unroll
    for (int r = 0; r < 8; ++r) {
        for (int m = 1; m < 16; m <<= 1) {
            pdot[r] += __shfl_xor(pdot[r], m, 32);
            pss[r]  += __shfl_xor(pss[r],  m, 32);
        }
    }
    if ((lane & 15) == 0) {
        long row0 = rowBase + (wave << 4) + ((lane >> 4) << 3);
#pragma unroll
        for (int r = 0; r < 8; ++r) {
            dotv[row0 + r] = pdot[r];
            ssv [row0 + r] = pss[r];
        }
    }
}

// ---------------------------------------------------------------------------
// Kernel 4: single-block softmax over W; writes relevance into d_out[H..H+W)
// also zeroes the r_mem accumulator for the following GEMV.
// ---------------------------------------------------------------------------
__global__ void __launch_bounds__(1024)
softmax_kernel(const float* __restrict__ dotv, const float* __restrict__ ssv,
               const float* __restrict__ qv, const int* __restrict__ mask,
               float* __restrict__ energy, float* __restrict__ rel_out,
               float* __restrict__ rmem, int W, int H) {
    __shared__ float red[32];
    const int tid = threadIdx.x, lane = tid & 31, wv = tid >> 5;

    for (int i = tid; i < H; i += 1024) rmem[i] = 0.f;

    // ||q||
    float p = 0.f;
    for (int i = tid; i < H; i += 1024) { float v = qv[i]; p += v * v; }
    for (int m = 1; m < 32; m <<= 1) p += __shfl_xor(p, m, 32);
    if (lane == 0) red[wv] = p;
    __syncthreads();
    if (wv == 0) {
        float t = red[lane];
        for (int m = 1; m < 32; m <<= 1) t += __shfl_xor(t, m, 32);
        if (lane == 0) red[0] = t;
    }
    __syncthreads();
    const float qn = sqrtf(red[0]);
    __syncthreads();

    // energy + max
    float mx = -3.4e38f;
    for (int i = tid; i < W; i += 1024) {
        float denom = fmaxf(sqrtf(ssv[i]) * qn, COS_EPS);
        float sim   = dotv[i] / denom;
        float dist  = __expf(-T_DIST * (float)(i + 1));
        float e     = (mask[i] == 0) ? MASK_FILL : sim * dist;
        energy[i] = e;
        mx = fmaxf(mx, e);
    }
    for (int m = 1; m < 32; m <<= 1) mx = fmaxf(mx, __shfl_xor(mx, m, 32));
    if (lane == 0) red[wv] = mx;
    __syncthreads();
    if (wv == 0) {
        float t = red[lane];
        for (int m = 1; m < 32; m <<= 1) t = fmaxf(t, __shfl_xor(t, m, 32));
        if (lane == 0) red[0] = t;
    }
    __syncthreads();
    mx = red[0];
    __syncthreads();

    // sum of exp
    float s = 0.f;
    for (int i = tid; i < W; i += 1024) s += __expf(energy[i] - mx);
    for (int m = 1; m < 32; m <<= 1) s += __shfl_xor(s, m, 32);
    if (lane == 0) red[wv] = s;
    __syncthreads();
    if (wv == 0) {
        float t = red[lane];
        for (int m = 1; m < 32; m <<= 1) t += __shfl_xor(t, m, 32);
        if (lane == 0) red[0] = t;
    }
    __syncthreads();
    const float inv = 1.f / red[0];

    for (int i = tid; i < W; i += 1024)
        rel_out[i] = __expf(energy[i] - mx) * inv;
}

// ---------------------------------------------------------------------------
// Kernel 5: r_mem[h] = sum_i relevance[i] * memory[i][h]  (bandwidth-bound)
// Each block sums 256 rows; coalesced column reads; atomicAdd partials.
// ---------------------------------------------------------------------------
__global__ void __launch_bounds__(256)
rmem_kernel(const float* __restrict__ mem, const float* __restrict__ rel,
            float* __restrict__ rmem, int W, int H) {
    const int tid = threadIdx.x;
    const long rb = (long)blockIdx.x * 256;
    for (int cbase = 0; cbase < H; cbase += 256) {
        float acc = 0.f;
        for (int r = 0; r < 256; ++r) {
            long row = rb + r;
            acc += rel[row] * mem[row * H + cbase + tid];
        }
        atomicAdd(&rmem[cbase + tid], acc);
    }
}

// ---------------------------------------------------------------------------
// Kernel 6: x[j] = sum_h r_mem[h] * Wv[j][h]
// ---------------------------------------------------------------------------
__global__ void xout_kernel(const float* __restrict__ rmem,
                            const float* __restrict__ Wv,
                            float* __restrict__ xo, int H) {
    int j = blockIdx.x * blockDim.x + threadIdx.x;
    if (j < H) {
        const float* wr = Wv + (size_t)j * H;
        float acc = 0.f;
        for (int h = 0; h < H; ++h) acc += rmem[h] * wr[h];
        xo[j] = acc;
    }
}

// ---------------------------------------------------------------------------
extern "C" void kernel_launch(void* const* d_in, const int* in_sizes, int n_in,
                              void* d_out, int out_size, void* d_ws, size_t ws_size,
                              hipStream_t stream) {
    const float* mem  = (const float*)d_in[0];
    const float* cu   = (const float*)d_in[1];
    const int*   mask = (const int*)d_in[2];
    const float* Wk   = (const float*)d_in[3];
    const float* Wq   = (const float*)d_in[4];
    const float* Wv   = (const float*)d_in[5];

    const int H = in_sizes[1];   // 768
    const int W = in_sizes[2];   // 65536

    float* out     = (float*)d_out;
    float* x_out   = out;        // [H]
    float* rel_out = out + H;    // [W]

    // workspace layout (256B-aligned regions)
    uint8_t* ws = (uint8_t*)d_ws;
    size_t off = 0;
    auto align256 = [](size_t v) { return (v + 255) & ~(size_t)255; };
    float* q = (float*)(ws + off);            off = align256(off + (size_t)H * 4);
    float* dotv = (float*)(ws + off);         off = align256(off + (size_t)W * 4);
    float* ssv = (float*)(ws + off);          off = align256(off + (size_t)W * 4);
    float* energy = (float*)(ws + off);       off = align256(off + (size_t)W * 4);
    float* rmem = (float*)(ws + off);         off = align256(off + (size_t)H * 4);
    unsigned short* WkT = (unsigned short*)(ws + off);
    // off += H*H*2  (~1.13 MB; total ws use ~2 MB)

    // 1) Wk bf16 convert (same layout, no transpose)
    wkt_convert_kernel<<<(H * H + 255) / 256, 256, 0, stream>>>(Wk, WkT, H);

    // 2) q projection
    qproj_kernel<<<(H + 255) / 256, 256, 0, stream>>>(cu, Wq, q, H);

    // 3) WMMA k-projection with fused dot/norm reductions
    size_t smem = (size_t)H * 4 + (size_t)128 * H * 2;  // q + bf16 A tile
    kproj_wmma_kernel<<<W / 128, 256, smem, stream>>>(mem, WkT, q, dotv, ssv, W, H);

    // 4) softmax -> relevance (d_out[H..]) ; zero rmem
    softmax_kernel<<<1, 1024, 0, stream>>>(dotv, ssv, q, mask, energy,
                                           rel_out, rmem, W, H);

    // 5) r_mem = relevance @ memory
    rmem_kernel<<<W / 256, 256, 0, stream>>>(mem, rel_out, rmem, W, H);

    // 6) x = r_mem @ Wv^T
    xout_kernel<<<(H + 255) / 256, 256, 0, stream>>>(rmem, Wv, x_out, H);
}
